// StageOne2_36670430773298
// MI455X (gfx1250) — compile-verified, matrix operand order checked by reading, not run on hardware
//
#include <hip/hip_runtime.h>
#include <hip/hip_bf16.h>
#include <math.h>

typedef __attribute__((ext_vector_type(16))) _Float16 v16h;
typedef __attribute__((ext_vector_type(8)))  _Float16 v8h;
typedef __attribute__((ext_vector_type(8)))  float    v8f;

#define BLK 256
#define BM  128
#define BN  128
#define BK  32
#define APITCH 40   // f16 row pitch (80B): 16B-aligned rows, conflict-dodging

// s_wait_asynccnt 0 (ASYNCcnt tracks global_load_async_to_lds_*)
#define ASYNC_WAIT0() asm volatile("s_wait_asynccnt 0x0" ::: "memory")

// One 16B lane-chunk: global (f16) -> LDS, no VGPR data, ASYNCcnt-tracked.
__device__ __forceinline__ void async_ld16(uint32_t lds_byte_off,
                                           const _Float16* gaddr) {
  asm volatile("global_load_async_to_lds_b128 %0, %1, off"
               :: "v"(lds_byte_off), "v"(gaddr) : "memory");
}

// Flat shared-pointer low 32 bits == LDS byte offset.
__device__ __forceinline__ uint32_t lds_off(const void* p) {
  return (uint32_t)(uintptr_t)p;
}

// ---------- block reductions ----------
__device__ __forceinline__ float blockReduceSum(float v, float* red) {
  int t = threadIdx.x;
  red[t] = v; __syncthreads();
  for (int off = blockDim.x >> 1; off > 0; off >>= 1) {
    if (t < off) red[t] += red[t + off];
    __syncthreads();
  }
  float r = red[0]; __syncthreads();
  return r;
}
__device__ __forceinline__ float blockReduceMax(float v, float* red) {
  int t = threadIdx.x;
  red[t] = v; __syncthreads();
  for (int off = blockDim.x >> 1; off > 0; off >>= 1) {
    if (t < off) red[t] = fmaxf(red[t], red[t + off]);
    __syncthreads();
  }
  float r = red[0]; __syncthreads();
  return r;
}

// 16x32 f16 operand fragment from an LDS tile row (A-layout, CDNA5 ISA):
// lane l holds row (l&15), K in {8h..8h+7} U {16+8h..16+8h+7}, h = l>>4.
__device__ __forceinline__ v16h ldsFrag(const _Float16* rowBase, int h) {
  v8h lo = *(const v8h*)(rowBase + h * 8);
  v8h hi = *(const v8h*)(rowBase + 16 + h * 8);
  v16h f;
#pragma unroll
  for (int e = 0; e < 8; e++) { f[e] = lo[e]; f[e + 8] = hi[e]; }
  return f;
}

// ---------- NT GEMM: C[M,N] = act(A[M,K] * B[N,K]^T + bias), f16 in, f32 acc.
// Tiles staged with double-buffered GLOBAL_LOAD_ASYNC_TO_LDS_B128.
// C (f32) and/or Ch (f16 copy) stores; fsq_out: sum(C^2) only (Frobenius^2).
// Requires K % 32 == 0 (inputs are pre-padded).
__global__ __launch_bounds__(256) void gemm_nt_wmma(
    const _Float16* __restrict__ A, const _Float16* __restrict__ Bw,
    const float* __restrict__ bias, float* __restrict__ C,
    _Float16* __restrict__ Ch, float* __restrict__ fsq_out,
    int M, int N, int K, int relu)
{
  (void)M;
  __shared__ _Float16 As[2][BM * APITCH];
  __shared__ _Float16 Bs[2][BN * APITCH];
  __shared__ float red[BLK];

  const int tid  = threadIdx.x;
  const int lane = tid & 31, wave = tid >> 5;
  const int wm = wave >> 1, wn = wave & 1;        // 4x2 wave grid: 32x64 each
  const int h = lane >> 4, ln = lane & 15;
  const int Mb = blockIdx.y * BM, Nb = blockIdx.x * BN;

  v8f acc[2][4];
  for (int i = 0; i < 2; i++)
    for (int j = 0; j < 4; j++)
      for (int e = 0; e < 8; e++) acc[i][j][e] = 0.0f;

  // issue async copies for one 128x32 f16 tile of A and B into buffer `buf`
  auto issue = [&](int buf, int k0) {
#pragma unroll
    for (int i = 0; i < 2; i++) {
      int idx = tid + BLK * i;                 // 512 chunks of 8 halves (16B)
      int r = idx >> 2, c8 = (idx & 3) << 3;
      async_ld16(lds_off(&As[buf][r * APITCH + c8]),
                 A + (size_t)(Mb + r) * K + k0 + c8);
      async_ld16(lds_off(&Bs[buf][r * APITCH + c8]),
                 Bw + (size_t)(Nb + r) * K + k0 + c8);
    }
  };

  issue(0, 0);
  int buf = 0;
  for (int k0 = 0; k0 < K; k0 += BK) {
    ASYNC_WAIT0();              // this wave's copies (incl. next-buf ones) done
    __syncthreads();            // all waves' copies visible in LDS
    if (k0 + BK < K) issue(buf ^ 1, k0 + BK);   // overlap copy with compute

    v16h af[2], bf[4];
#pragma unroll
    for (int mt = 0; mt < 2; mt++)
      af[mt] = ldsFrag(&As[buf][(wm * 32 + mt * 16 + ln) * APITCH], h);
#pragma unroll
    for (int nt = 0; nt < 4; nt++)
      bf[nt] = ldsFrag(&Bs[buf][(wn * 64 + nt * 16 + ln) * APITCH], h);
#pragma unroll
    for (int mt = 0; mt < 2; mt++)
#pragma unroll
      for (int nt = 0; nt < 4; nt++)
        acc[mt][nt] = __builtin_amdgcn_wmma_f32_16x16x32_f16(
            false, af[mt], false, bf[nt], (short)0, acc[mt][nt], false, false);

    __syncthreads();            // readers done before this buffer is refilled
    buf ^= 1;
  }

  if (fsq_out) {
    float s = 0.0f;
    for (int mt = 0; mt < 2; mt++)
      for (int nt = 0; nt < 4; nt++)
        for (int e = 0; e < 8; e++) { float v = acc[mt][nt][e]; s += v * v; }
    s = blockReduceSum(s, red);
    if (tid == 0) atomicAdd(fsq_out, s);
  } else {
    for (int mt = 0; mt < 2; mt++)
      for (int nt = 0; nt < 4; nt++) {
        int n = Nb + wn * 64 + nt * 16 + ln;
        float bv = bias ? bias[n] : 0.0f;
        for (int e = 0; e < 8; e++) {
          int m = Mb + wm * 32 + mt * 16 + h * 8 + e;
          float v = acc[mt][nt][e] + bv;
          if (relu) v = fmaxf(v, 0.0f);
          if (C)  C[(size_t)m * N + n] = v;
          if (Ch) Ch[(size_t)m * N + n] = (_Float16)v;
        }
      }
  }
}

// ---------- small kernels ----------
// f32 [R,C] -> f16 [R,Cp], zero-padded columns (makes every GEMM K % 32 == 0)
__global__ void cvt_pad_f16(const float* __restrict__ X, _Float16* __restrict__ Y,
                            int R, int C, int Cp) {
  size_t i = (size_t)blockIdx.x * blockDim.x + threadIdx.x;
  if (i >= (size_t)R * Cp) return;
  int r = (int)(i / Cp), c = (int)(i % Cp);
  Y[i] = (c < C) ? (_Float16)X[(size_t)r * C + c] : (_Float16)0.0f;
}

__global__ void zero_scalars(float* s, int* n0) {
  s[0] = 0.f; s[1] = 0.f; s[2] = 0.f; s[3] = 0.f; *n0 = 0;
}

__global__ void county(const int* __restrict__ y, int B, int* __restrict__ n0) {
  __shared__ int red[256];
  int j = blockIdx.x * blockDim.x + threadIdx.x;
  red[threadIdx.x] = (j < B && y[j] == 0) ? 1 : 0;
  __syncthreads();
  for (int off = 128; off > 0; off >>= 1) {
    if ((int)threadIdx.x < off) red[threadIdx.x] += red[threadIdx.x + off];
    __syncthreads();
  }
  if (threadIdx.x == 0) atomicAdd(n0, red[0]);
}

// row-wise L2 normalize of f32 [B,128] -> f16 (feeds gram GEMMs only)
__global__ __launch_bounds__(128) void rownorm(const float* __restrict__ X,
                                               _Float16* __restrict__ Y) {
  __shared__ float red[128];
  int r = blockIdx.x, t = threadIdx.x;
  float v = X[(size_t)r * 128 + t];
  float ss = blockReduceSum(v * v, red);
  float nrm = fmaxf(sqrtf(ss), 1e-12f);
  Y[(size_t)r * 128 + t] = (_Float16)(v / nrm);
}

// Per-row softmax + symmetric KL vs the two-valued label distribution q.
__global__ __launch_bounds__(256) void clip_kl(
    const float* __restrict__ G, const int* __restrict__ y,
    const int* __restrict__ n0p, const float* __restrict__ logscale,
    float* __restrict__ acc, int B)
{
  __shared__ float red[256];
  const int PT = 16;                       // 4096 / 256
  int k = blockIdx.x, t = threadIdx.x;
  float scale = (logscale != nullptr) ? __expf(*logscale) : 1.0f;
  int yk = y[k];
  const float* row = G + (size_t)k * B;
  float g[PT]; int yj[PT];
#pragma unroll
  for (int i = 0; i < PT; i++) { int j = i * 256 + t; g[i] = row[j]; yj[i] = y[j]; }

  float m = -3.0e38f;
#pragma unroll
  for (int i = 0; i < PT; i++) m = fmaxf(m, g[i]);
  m = blockReduceMax(m, red);

  float se = 0.0f;
#pragma unroll
  for (int i = 0; i < PT; i++) se += __expf(scale * (g[i] - m));
  se = blockReduceSum(se, red);
  float lse = scale * m + __logf(se);

  int n0 = *n0p;
  int sc = (yk == 0) ? n0 : (B - n0);
  float inv = 1.0f / (float)sc;
  float Z   = (float)sc * __expf(inv) + (float)(B - sc);
  float lZ  = __logf(Z);
  float lqm = inv - lZ, lqn = -lZ;
  float qm  = __expf(lqm), qn = __expf(lqn);

  float local = 0.0f;
#pragma unroll
  for (int i = 0; i < PT; i++) {
    float lp = scale * g[i] - lse;
    float p  = __expf(lp);
    bool mt  = (yj[i] == yk);
    float q  = mt ? qm : qn;
    float lq = mt ? lqm : lqn;
    local += q * (lq - lp) + p * (lp - lq);
  }
  local = blockReduceSum(local, red);
  if (t == 0) atomicAdd(acc, local);
}

// 4 classifier heads
__global__ __launch_bounds__(128) void heads(
    const float* __restrict__ ISP, const float* __restrict__ ISH,
    const float* __restrict__ TSP, const float* __restrict__ TSH,
    const float* __restrict__ c1w, const float* __restrict__ c1b,
    const float* __restrict__ c2w, const float* __restrict__ c2b,
    const float* __restrict__ c3w, const float* __restrict__ c3b,
    const float* __restrict__ c4w, const float* __restrict__ c4b,
    float* __restrict__ out, int B)
{
  __shared__ float red[128];
  int r = blockIdx.x, t = threadIdx.x;
  size_t o = (size_t)r * 128 + t;
  float d1 = blockReduceSum(ISP[o] * c1w[t], red);
  float d2 = blockReduceSum(ISH[o] * c2w[t], red);
  float d4 = blockReduceSum(TSP[o] * c4w[t], red);
  float d3 = blockReduceSum(TSH[o] * c3w[t], red);
  if (t == 0) {
    out[1 + r]         = d1 + c1b[0];   // i_sp_out
    out[1 + B + r]     = d2 + c2b[0];   // i_sh_out
    out[1 + 2 * B + r] = d4 + c4b[0];   // t_sp_out
    out[1 + 3 * B + r] = d3 + c3b[0];   // t_sh_out
  }
}

__global__ void final_loss(const float* __restrict__ s, float* __restrict__ out, int B) {
  float loss1   = s[0] / (4.0f * (float)B);
  float loss_sh = s[1] / (4.0f * (float)B);
  float loss_sp = 0.5f * (sqrtf(s[2]) + sqrtf(s[3]));
  out[0] = (loss_sp + loss_sh + loss1) / 3.0f;
}

// ---------- launcher ----------
extern "C" void kernel_launch(void* const* d_in, const int* in_sizes, int n_in,
                              void* d_out, int out_size, void* d_ws, size_t ws_size,
                              hipStream_t stream) {
  (void)in_sizes; (void)n_in; (void)out_size; (void)ws_size;
  const float* I    = (const float*)d_in[0];
  const float* T    = (const float*)d_in[1];
  const int*   Y    = (const int*)d_in[2];
  const float* l1w  = (const float*)d_in[3];  const float* l1b = (const float*)d_in[4];
  const float* l2w  = (const float*)d_in[5];  const float* l2b = (const float*)d_in[6];
  const float* l3w  = (const float*)d_in[7];  const float* l3b = (const float*)d_in[8];
  const float* tl1w = (const float*)d_in[9];  const float* tl1b= (const float*)d_in[10];
  const float* tl2w = (const float*)d_in[11]; const float* tl2b= (const float*)d_in[12];
  const float* ispw = (const float*)d_in[13]; const float* ispb= (const float*)d_in[14];
  const float* ishw = (const float*)d_in[15]; const float* ishb= (const float*)d_in[16];
  const float* tshw = (const float*)d_in[17]; const float* tshb= (const float*)d_in[18];
  const float* tspw = (const float*)d_in[19]; const float* tspb= (const float*)d_in[20];
  const float* c1w  = (const float*)d_in[21]; const float* c1b = (const float*)d_in[22];
  const float* c2w  = (const float*)d_in[23]; const float* c2b = (const float*)d_in[24];
  const float* c3w  = (const float*)d_in[25]; const float* c3b = (const float*)d_in[26];
  const float* c4w  = (const float*)d_in[27]; const float* c4b = (const float*)d_in[28];
  const float* scale1 = (const float*)d_in[29];
  const float* scale2 = (const float*)d_in[30];
  float* out = (float*)d_out;

  const int B = 4096;
  const size_t F = (size_t)B * 128;

  // f32 region
  float* ws = (float*)d_ws;
  size_t o = 0;
  float* G   = ws + o; o += (size_t)B * B;
  float* FI  = ws + o; o += F;
  float* FT  = ws + o; o += F;
  float* ISP = ws + o; o += F;
  float* ISH = ws + o; o += F;
  float* TSH = ws + o; o += F;
  float* TSP = ws + o; o += F;
  float* scal= ws + o; o += 8;                    // [kl1, kl2, fsq_t, fsq_i]
  int*   n0  = (int*)(scal + 4);

  // f16 region
  _Float16* hbase = (_Float16*)(ws + o);
  size_t ho = 0;
  _Float16* I16   = hbase + ho; ho += (size_t)B * 2048;
  _Float16* Tp16  = hbase + ho; ho += (size_t)B * 64;     // T padded 49 -> 64
  _Float16* l1w16 = hbase + ho; ho += (size_t)512 * 2048;
  _Float16* l2w16 = hbase + ho; ho += (size_t)256 * 512;
  _Float16* l3w16 = hbase + ho; ho += (size_t)128 * 256;
  _Float16* tl1w16= hbase + ho; ho += (size_t)128 * 64;   // padded 49 -> 64
  _Float16* tl2w16= hbase + ho; ho += (size_t)128 * 128;
  _Float16* ispw16= hbase + ho; ho += (size_t)128 * 128;
  _Float16* ishw16= hbase + ho; ho += (size_t)128 * 128;
  _Float16* tshw16= hbase + ho; ho += (size_t)128 * 128;
  _Float16* tspw16= hbase + ho; ho += (size_t)128 * 128;
  _Float16* H1_16 = hbase + ho; ho += (size_t)B * 512;
  _Float16* H2_16 = hbase + ho; ho += (size_t)B * 256;
  _Float16* T1_16 = hbase + ho; ho += F;
  _Float16* FI16  = hbase + ho; ho += F;
  _Float16* FT16  = hbase + ho; ho += F;
  _Float16* ISP16 = hbase + ho; ho += F;
  _Float16* ISH16 = hbase + ho; ho += F;
  _Float16* TSH16 = hbase + ho; ho += F;
  _Float16* TSP16 = hbase + ho; ho += F;
  _Float16* FIn16 = hbase + ho; ho += F;
  _Float16* FTn16 = hbase + ho; ho += F;
  _Float16* ISHn16= hbase + ho; ho += F;
  _Float16* TSHn16= hbase + ho; ho += F;

  zero_scalars<<<1, 1, 0, stream>>>(scal, n0);
  county<<<B / 256, 256, 0, stream>>>(Y, B, n0);

  auto cvt = [&](const float* X, _Float16* Yh, int R, int Cc, int Cp) {
    size_t total = (size_t)R * Cp;
    cvt_pad_f16<<<(int)((total + 255) / 256), 256, 0, stream>>>(X, Yh, R, Cc, Cp);
  };
  cvt(I, I16, B, 2048, 2048);
  cvt(T, Tp16, B, 49, 64);
  cvt(l1w, l1w16, 512, 2048, 2048);
  cvt(l2w, l2w16, 256, 512, 512);
  cvt(l3w, l3w16, 128, 256, 256);
  cvt(tl1w, tl1w16, 128, 49, 64);
  cvt(tl2w, tl2w16, 128, 128, 128);
  cvt(ispw, ispw16, 128, 128, 128);
  cvt(ishw, ishw16, 128, 128, 128);
  cvt(tshw, tshw16, 128, 128, 128);
  cvt(tspw, tspw16, 128, 128, 128);

  dim3 blk(256);
  auto gemm = [&](const _Float16* A, const _Float16* Bw, const float* bias,
                  float* C, _Float16* Ch, float* fsq, int M, int N, int K, int relu) {
    dim3 grid(N / 128, M / 128);
    gemm_nt_wmma<<<grid, blk, 0, stream>>>(A, Bw, bias, C, Ch, fsq, M, N, K, relu);
  };

  // encoders (f16 chain; K always % 32 == 0 after padding)
  gemm(I16,   l1w16,  l1b,  nullptr, H1_16, nullptr, B, 512, 2048, 1);
  gemm(H1_16, l2w16,  l2b,  nullptr, H2_16, nullptr, B, 256, 512, 1);
  gemm(H2_16, l3w16,  l3b,  FI,      FI16,  nullptr, B, 128, 256, 0);
  gemm(Tp16,  tl1w16, tl1b, nullptr, T1_16, nullptr, B, 128, 64, 1);
  gemm(T1_16, tl2w16, tl2b, FT,      FT16,  nullptr, B, 128, 128, 0);

  // loss1: clip(fi, ft, exp(scale1))
  rownorm<<<B, 128, 0, stream>>>(FI, FIn16);
  rownorm<<<B, 128, 0, stream>>>(FT, FTn16);
  gemm(FIn16, FTn16, nullptr, G, nullptr, nullptr, B, B, 128, 0);
  clip_kl<<<B, 256, 0, stream>>>(G, Y, n0, scale1, &scal[0], B);
  gemm(FTn16, FIn16, nullptr, G, nullptr, nullptr, B, B, 128, 0);
  clip_kl<<<B, 256, 0, stream>>>(G, Y, n0, nullptr, &scal[0], B);

  // SS heads
  gemm(FI16, ispw16, ispb, ISP, ISP16, nullptr, B, 128, 128, 0);
  gemm(FI16, ishw16, ishb, ISH, ISH16, nullptr, B, 128, 128, 0);
  gemm(FT16, tshw16, tshb, TSH, TSH16, nullptr, B, 128, 128, 0);
  gemm(FT16, tspw16, tspb, TSP, TSP16, nullptr, B, 128, 128, 0);

  // loss_sh: clip(i_sh, t_sh, exp(scale2))
  rownorm<<<B, 128, 0, stream>>>(ISH, ISHn16);
  rownorm<<<B, 128, 0, stream>>>(TSH, TSHn16);
  gemm(ISHn16, TSHn16, nullptr, G, nullptr, nullptr, B, B, 128, 0);
  clip_kl<<<B, 256, 0, stream>>>(G, Y, n0, scale2, &scal[1], B);
  gemm(TSHn16, ISHn16, nullptr, G, nullptr, nullptr, B, B, 128, 0);
  clip_kl<<<B, 256, 0, stream>>>(G, Y, n0, nullptr, &scal[1], B);

  // loss_sp via Frobenius^2 epilogue (never materialized)
  gemm(TSP16, TSH16, nullptr, nullptr, nullptr, &scal[2], B, B, 128, 0);
  gemm(ISP16, ISH16, nullptr, nullptr, nullptr, &scal[3], B, B, 128, 0);

  // outputs
  heads<<<B, 128, 0, stream>>>(ISP, ISH, TSP, TSH, c1w, c1b, c2w, c2b,
                               c3w, c3b, c4w, c4b, out, B);
  final_loss<<<1, 1, 0, stream>>>(scal, out, B);
}
